// ULiteModel_Continuous_RegSeg_26302379720996
// MI455X (gfx1250) — matrix operand order repaired and checked by reading, not compiled, not run on hardware
//
#include <hip/hip_runtime.h>
#include <math.h>

typedef __attribute__((ext_vector_type(16))) _Float16 v16h;
typedef __attribute__((ext_vector_type(8)))  float    v8f;

#define EPSV 1e-5f
#define M_BN    1
#define M_PRELU 2
#define M_SIG   4
#define M_BIAS  8

#define NB 4          // batch
#define KCHUNK 32
#define TILE_PX 256   // 8 waves * 32 px each
#define TILE_CO 32    // 2 wmma tiles of 16 rows

__device__ __forceinline__ float sigm(float x){ return 1.f/(1.f+expf(-x)); }

// ---------------------------------------------------------------------------
// WMMA 1x1 conv (GEMM): Y[n,co,p] = sum_ci W[co,ci]*X[n,ci,p], fused epilogue.
// Block tile 32co x 256px; each wave owns a 32co x 32px tile = 4 accumulators,
// issuing 4 back-to-back v_wmma per K-chunk with 2 A- and 2 B-fragments.
// X addr = ((n*Cin+ci)*XHW)+xoff+p, p in [0,HW); Y analogous.
// ---------------------------------------------------------------------------
__global__ __launch_bounds__(256) void k_conv1x1(
    const float* __restrict__ X, int XHW, int xoff,
    const float* __restrict__ W,
    const float* __restrict__ bng, const float* __restrict__ bnb,
    const float* __restrict__ bnm, const float* __restrict__ bnv,
    const float* __restrict__ pa,  const float* __restrict__ bias,
    float* __restrict__ Y, int YHW, int yoff,
    int Cin, int Cout, int HW, int Npix, int mode)
{
  __shared__ _Float16 Xs[TILE_PX][KCHUNK];   // [px][k] 16KB
  __shared__ _Float16 Ws[TILE_CO][KCHUNK];   // [co][k]  2KB
  const int tid  = threadIdx.x;
  const int wave = tid >> 5;
  const int lane = tid & 31;
  const int q0   = blockIdx.x * TILE_PX;
  const int co0  = blockIdx.y * TILE_CO;
  const int hi   = (lane >> 4) & 1;
  const int lrow = lane & 15;

  v8f acc00 = {}, acc01 = {}, acc10 = {}, acc11 = {};

  for (int kc = 0; kc < Cin; kc += KCHUNK) {
    // stage X tile (f32 -> f16): 256x32 elements, 32 per thread
    #pragma unroll 4
    for (int pass = 0; pass < (TILE_PX*KCHUNK)/256; ++pass) {
      int i  = pass*256 + tid;
      int k  = i >> 8;          // / TILE_PX
      int px = i & (TILE_PX-1);
      int q  = q0 + px;
      int kk = kc + k;
      float val = 0.f;
      if (q < Npix && kk < Cin) {
        int n = q / HW, p = q - n*HW;
        val = X[((long)n*Cin + kk)*XHW + xoff + p];
      }
      Xs[px][k] = (_Float16)val;
    }
    // stage W tile: 32x32 elements, 4 per thread
    #pragma unroll
    for (int pass = 0; pass < (TILE_CO*KCHUNK)/256; ++pass) {
      int i  = pass*256 + tid;
      int co = i >> 5, k = i & 31;
      int kk = kc + k, c = co0 + co;
      float val = (c < Cout && kk < Cin) ? W[(long)c*Cin + kk] : 0.f;
      Ws[co][k] = (_Float16)val;
    }
    __syncthreads();

    // A fragments (16x32 each): row = lane&15 (+16 for 2nd co tile),
    // K packing per documented 16-bit A layout.
    v16h a0, a1, b0, b1;
    #pragma unroll
    for (int j = 0; j < 16; ++j) {
      int vv = j >> 1;
      int kb = (vv < 4) ? (vv*2) : (16 + (vv-4)*2);
      int ka = kb + (j & 1) + hi*8;
      a0[j] = Ws[lrow][ka];
      a1[j] = Ws[16 + lrow][ka];
    }
    // B fragments (32x16 each): col = lane&15 within px subtile, K = hi*8+j (+8 for j>=8)
    const int colb = wave*32 + lrow;
    const int bk   = hi*8;
    #pragma unroll
    for (int j = 0; j < 16; ++j) {
      b0[j] = Xs[colb][bk + j + ((j >= 8) ? 8 : 0)];
      b1[j] = Xs[colb + 16][bk + j + ((j >= 8) ? 8 : 0)];
    }

    acc00 = __builtin_amdgcn_wmma_f32_16x16x32_f16(false, a0, false, b0, (short)0, acc00, false, false);
    acc01 = __builtin_amdgcn_wmma_f32_16x16x32_f16(false, a0, false, b1, (short)0, acc01, false, false);
    acc10 = __builtin_amdgcn_wmma_f32_16x16x32_f16(false, a1, false, b0, (short)0, acc10, false, false);
    acc11 = __builtin_amdgcn_wmma_f32_16x16x32_f16(false, a1, false, b1, (short)0, acc11, false, false);

    // prefetch next K chunk of X into cache
    if (kc + KCHUNK < Cin) {
      int kk = kc + KCHUNK + (tid >> 6)*8;
      int q  = q0 + (tid & 63)*4;
      if (q < Npix && kk < Cin) {
        int n = q / HW, p = q - n*HW;
        __builtin_prefetch(&X[((long)n*Cin + kk)*XHW + xoff + p], 0, 1);
      }
    }
    __syncthreads();
  }

  // epilogue: C/D layout -> co = cobase + hi*8 + r, pixel col = lane&15
  #pragma unroll
  for (int t = 0; t < 4; ++t) {
    const v8f& acc = (t == 0) ? acc00 : (t == 1) ? acc01 : (t == 2) ? acc10 : acc11;
    const int cot  = (t >> 1);          // co tile 0/1
    const int pxt  = (t & 1);           // px subtile 0/1
    const int q = q0 + wave*32 + pxt*16 + lrow;
    if (q >= Npix) continue;
    const int n = q / HW, p = q - n*HW;
    #pragma unroll
    for (int r = 0; r < 8; ++r) {
      int co = co0 + cot*16 + hi*8 + r;
      if (co >= Cout) continue;
      float val = acc[r];
      if (mode & M_BIAS) val += bias[co];
      if (mode & M_BN) {
        float sc = bng[co] * rsqrtf(bnv[co] + EPSV);
        val = (val - bnm[co]) * sc + bnb[co];
      }
      if (mode & M_PRELU) val = (val >= 0.f) ? val : pa[co]*val;
      if (mode & M_SIG)   val = sigm(val);
      Y[((long)n*Cout + co)*YHW + yoff + p] = val;
    }
  }
}

// ---------------------------------------------------------------------------
// input 3x3 conv (3->16) + bias
// ---------------------------------------------------------------------------
__global__ void k_conv3x3_in(const float* __restrict__ X, const float* __restrict__ Wt,
                             const float* __restrict__ Bs, float* __restrict__ Y,
                             int N, int H, int W)
{
  long idx = (long)blockIdx.x*blockDim.x + threadIdx.x;
  long total = (long)N*16*H*W;
  if (idx >= total) return;
  int w = idx % W; int h = (idx / W) % H;
  int co = (idx / ((long)W*H)) % 16; int n = idx / ((long)16*H*W);
  float s = Bs[co];
  for (int ci = 0; ci < 3; ++ci)
    for (int dy = -1; dy <= 1; ++dy) {
      int y = h + dy; if ((unsigned)y >= (unsigned)H) continue;
      for (int dx = -1; dx <= 1; ++dx) {
        int x = w + dx; if ((unsigned)x >= (unsigned)W) continue;
        s += X[((long)(n*3+ci)*H + y)*W + x] * Wt[((co*3+ci)*3 + (dy+1))*3 + (dx+1)];
      }
    }
  Y[idx] = s;
}

// ---------------------------------------------------------------------------
// fused cadw(m) + cadw(l): S = 2x + sum taps (fractional shifts H and W)
// ---------------------------------------------------------------------------
__device__ __forceinline__ float fshiftH(const float* xc, int h, int w, float s, int H, int W){
  float fs = floorf(s); int d0 = (int)fs; float f = s - fs;
  int i0 = h + d0, i1 = i0 + 1;
  float v0 = ((unsigned)i0 < (unsigned)H) ? xc[(long)i0*W + w] : 0.f;
  float v1 = ((unsigned)i1 < (unsigned)H) ? xc[(long)i1*W + w] : 0.f;
  return v0*(1.f-f) + v1*f;
}
__device__ __forceinline__ float fshiftW(const float* xc, int h, int w, float s, int W){
  float fs = floorf(s); int d0 = (int)fs; float f = s - fs;
  int i0 = w + d0, i1 = i0 + 1;
  float v0 = ((unsigned)i0 < (unsigned)W) ? xc[(long)h*W + i0] : 0.f;
  float v1 = ((unsigned)i1 < (unsigned)W) ? xc[(long)h*W + i1] : 0.f;
  return v0*(1.f-f) + v1*f;
}
__global__ void k_cadw2(const float* __restrict__ X,
                        const float* __restrict__ rm_p, const float* __restrict__ whm, const float* __restrict__ wwm,
                        const float* __restrict__ rl_p, const float* __restrict__ whl, const float* __restrict__ wwl,
                        float* __restrict__ S, int N, int C, int H, int W, int K)
{
  long idx = (long)blockIdx.x*blockDim.x + threadIdx.x;
  long total = (long)N*C*H*W;
  if (idx >= total) return;
  int w = idx % W; int h = (idx / W) % H;
  int c = (idx / ((long)W*H)) % C; int n = idx / ((long)C*H*W);
  float rm = fmaxf(rm_p[0], 1.f), rl = fmaxf(rl_p[0], 1.f);
  const float* xc = X + (long)(n*C + c)*H*W;
  float acc = 2.f * xc[(long)h*W + w];
  int half = K/2;
  for (int i = 0; i < K; ++i) {
    float off = (float)(i - half);
    acc += fshiftH(xc, h, w, off*rm, H, W) * whm[c*K + i];
    acc += fshiftW(xc, h, w, off*rm, W)    * wwm[c*K + i];
    acc += fshiftH(xc, h, w, off*rl, H, W) * whl[c*K + i];
    acc += fshiftW(xc, h, w, off*rl, W)    * wwl[c*K + i];
  }
  S[idx] = acc;
}

// ---------------------------------------------------------------------------
// detail-guidance + residual + prelu:  O = prelu(F + bn(x + dw3x1 + dw1x3))
// ---------------------------------------------------------------------------
__global__ void k_dg_add_prelu(const float* __restrict__ X, const float* __restrict__ F,
                               const float* __restrict__ wh, const float* __restrict__ ww,
                               const float* __restrict__ g, const float* __restrict__ b,
                               const float* __restrict__ m, const float* __restrict__ v,
                               const float* __restrict__ act, float* __restrict__ O,
                               int N, int C, int H, int W)
{
  long idx = (long)blockIdx.x*blockDim.x + threadIdx.x;
  long total = (long)N*C*H*W;
  if (idx >= total) return;
  int w = idx % W; int h = (idx / W) % H;
  int c = (idx / ((long)W*H)) % C; int n = idx / ((long)C*H*W);
  const float* xc = X + (long)(n*C + c)*H*W;
  float e = 0.f;
  for (int i = 0; i < 3; ++i) {
    int y = h + i - 1;
    if ((unsigned)y < (unsigned)H) e += xc[(long)y*W + w] * wh[c*3 + i];
    int x = w + i - 1;
    if ((unsigned)x < (unsigned)W) e += xc[(long)h*W + x] * ww[c*3 + i];
  }
  float val = xc[(long)h*W + w] + e;
  float sc = g[c] * rsqrtf(v[c] + EPSV);
  val = (val - m[c]) * sc + b[c];
  val = F[idx] + val;
  O[idx] = (val >= 0.f) ? val : act[c]*val;
}

// ---------------------------------------------------------------------------
// coord attention helpers
// ---------------------------------------------------------------------------
__global__ void k_ca_pool(const float* __restrict__ X, float* __restrict__ Yc,
                          int N, int C, int H, int W)
{
  long idx = (long)blockIdx.x*blockDim.x + threadIdx.x;
  int L = H + W;
  long total = (long)N*C*L;
  if (idx >= total) return;
  int pos = idx % L; int c = (idx / L) % C; int n = idx / ((long)C*L);
  const float* xc = X + (long)(n*C + c)*H*W;
  float s = 0.f;
  if (pos < H) { for (int w = 0; w < W; ++w) s += xc[(long)pos*W + w]; s /= (float)W; }
  else { int ww = pos - H; for (int h = 0; h < H; ++h) s += xc[(long)h*W + ww]; s /= (float)H; }
  Yc[idx] = s;
}
__global__ void k_ca_apply(const float* __restrict__ X, const float* __restrict__ Ah,
                           const float* __restrict__ Aw, float* __restrict__ Y,
                           int N, int C, int H, int W)
{
  long idx = (long)blockIdx.x*blockDim.x + threadIdx.x;
  long total = (long)N*C*H*W;
  if (idx >= total) return;
  int w = idx % W; int h = (idx / W) % H;
  int c = (idx / ((long)W*H)) % C; int n = idx / ((long)C*H*W);
  Y[idx] = X[idx] * Aw[(long)(n*C + c)*W + w] * Ah[(long)(n*C + c)*H + h];
}

// ---------------------------------------------------------------------------
// elementwise bn / prelu
// ---------------------------------------------------------------------------
__global__ void k_bn_act(const float* __restrict__ X, float* __restrict__ Y,
                         const float* __restrict__ g, const float* __restrict__ b,
                         const float* __restrict__ m, const float* __restrict__ v,
                         const float* __restrict__ a, int C, int HW, long total, int mode)
{
  long idx = (long)blockIdx.x*blockDim.x + threadIdx.x;
  if (idx >= total) return;
  int c = (idx / HW) % C;
  float val = X[idx];
  if (mode & M_BN) {
    float sc = g[c] * rsqrtf(v[c] + EPSV);
    val = (val - m[c]) * sc + b[c];
  }
  if (mode & M_PRELU) val = (val >= 0.f) ? val : a[c]*val;
  Y[idx] = val;
}

// ---------------------------------------------------------------------------
// maxpool 2x2 (writes into concat buffer at channel offset)
// ---------------------------------------------------------------------------
__global__ void k_maxpool2(const float* __restrict__ X, float* __restrict__ Y,
                           int N, int C, int H, int W, int CoutTot, int coff)
{
  int Ho = H/2, Wo = W/2;
  long idx = (long)blockIdx.x*blockDim.x + threadIdx.x;
  long total = (long)N*C*Ho*Wo;
  if (idx >= total) return;
  int wo = idx % Wo; int ho = (idx / Wo) % Ho;
  int c = (idx / ((long)Wo*Ho)) % C; int n = idx / ((long)C*Ho*Wo);
  const float* xc = X + (long)(n*C + c)*H*W;
  int h = ho*2, w = wo*2;
  float v0 = xc[(long)h*W + w],     v1 = xc[(long)h*W + w + 1];
  float v2 = xc[(long)(h+1)*W + w], v3 = xc[(long)(h+1)*W + w + 1];
  float mx = fmaxf(fmaxf(v0, v1), fmaxf(v2, v3));
  Y[((long)(n*CoutTot + coff + c)*Ho + ho)*Wo + wo] = mx;
}

// ---------------------------------------------------------------------------
// adaptive average pool to o x o
// ---------------------------------------------------------------------------
__global__ void k_adaptive_avg(const float* __restrict__ X, float* __restrict__ Y,
                               int N, int C, int H, int W, int o)
{
  long idx = (long)blockIdx.x*blockDim.x + threadIdx.x;
  long total = (long)N*C*o*o;
  if (idx >= total) return;
  int ox = idx % o; int oy = (idx / o) % o;
  int c = (idx / (o*o)) % C; int n = idx / ((long)C*o*o);
  int hb = H/o, wb = W/o;
  const float* xc = X + (long)(n*C + c)*H*W;
  float s = 0.f;
  for (int y = oy*hb; y < (oy+1)*hb; ++y)
    for (int x = ox*wb; x < (ox+1)*wb; ++x) s += xc[(long)y*W + x];
  Y[idx] = s / (float)(hb*wb);
}

// ---------------------------------------------------------------------------
// bilinear resize (half-pixel centers), writes at channel offset
// ---------------------------------------------------------------------------
__global__ void k_resize(const float* __restrict__ X, int N, int C, int Hi, int Wi,
                         float* __restrict__ Y, int Ho, int Wo, int CoutTot, int coff)
{
  long idx = (long)blockIdx.x*blockDim.x + threadIdx.x;
  long total = (long)N*C*Ho*Wo;
  if (idx >= total) return;
  int x = idx % Wo; int y = (idx / Wo) % Ho;
  int c = (idx / ((long)Wo*Ho)) % C; int n = idx / ((long)C*Ho*Wo);
  float sy = (float)Hi/(float)Ho, sx = (float)Wi/(float)Wo;
  float fy = (y + 0.5f)*sy - 0.5f, fx = (x + 0.5f)*sx - 0.5f;
  float y0f = floorf(fy), x0f = floorf(fx);
  float wy = fy - y0f, wx = fx - x0f;
  int y0 = (int)y0f, x0 = (int)x0f;
  int y0c = y0 < 0 ? 0 : (y0 > Hi-1 ? Hi-1 : y0);
  int y1c = y0+1 < 0 ? 0 : (y0+1 > Hi-1 ? Hi-1 : y0+1);
  int x0c = x0 < 0 ? 0 : (x0 > Wi-1 ? Wi-1 : x0);
  int x1c = x0+1 < 0 ? 0 : (x0+1 > Wi-1 ? Wi-1 : x0+1);
  const float* xc = X + (long)(n*C + c)*Hi*Wi;
  float v = (1.f-wy)*((1.f-wx)*xc[(long)y0c*Wi + x0c] + wx*xc[(long)y0c*Wi + x1c])
          +      wy *((1.f-wx)*xc[(long)y1c*Wi + x0c] + wx*xc[(long)y1c*Wi + x1c]);
  Y[((long)(n*CoutTot + coff + c)*Ho + y)*Wo + x] = v;
}

// ---------------------------------------------------------------------------
// channel-slab copy into concat buffer
// ---------------------------------------------------------------------------
__global__ void k_copy_ch(const float* __restrict__ X, float* __restrict__ Y,
                          int N, int C, int HW, int CoutTot, int coff)
{
  long idx = (long)blockIdx.x*blockDim.x + threadIdx.x;
  long total = (long)N*C*HW;
  if (idx >= total) return;
  int p = idx % HW; int c = (idx / HW) % C; int n = idx / ((long)C*HW);
  Y[((long)(n*CoutTot + coff + c))*HW + p] = X[idx];
}

// ---------------------------------------------------------------------------
// depthwise 5x5 + bn + residual:  O = bn(dw5(F)) + F
// ---------------------------------------------------------------------------
__global__ void k_dw5_bn_add(const float* __restrict__ F, const float* __restrict__ Wd,
                             const float* __restrict__ g, const float* __restrict__ b,
                             const float* __restrict__ m, const float* __restrict__ v,
                             float* __restrict__ O, int N, int C, int H, int W)
{
  long idx = (long)blockIdx.x*blockDim.x + threadIdx.x;
  long total = (long)N*C*H*W;
  if (idx >= total) return;
  int w = idx % W; int h = (idx / W) % H;
  int c = (idx / ((long)W*H)) % C; int n = idx / ((long)C*H*W);
  const float* fc = F + (long)(n*C + c)*H*W;
  float e = 0.f;
  for (int dy = -2; dy <= 2; ++dy) {
    int y = h + dy; if ((unsigned)y >= (unsigned)H) continue;
    for (int dx = -2; dx <= 2; ++dx) {
      int x = w + dx; if ((unsigned)x >= (unsigned)W) continue;
      e += fc[(long)y*W + x] * Wd[c*25 + (dy+2)*5 + (dx+2)];
    }
  }
  float sc = g[c] * rsqrtf(v[c] + EPSV);
  O[idx] = (e - m[c])*sc + b[c] + F[idx];
}

// ---------------------------------------------------------------------------
// UAFM helpers
// ---------------------------------------------------------------------------
__global__ void k_chan_reduce(const float* __restrict__ A, const float* __restrict__ B,
                              float* __restrict__ Si, int N, int C, int HW)
{
  long idx = (long)blockIdx.x*blockDim.x + threadIdx.x;
  long total = (long)N*HW;
  if (idx >= total) return;
  int p = idx % HW; int n = idx / HW;
  float s = 0.f, mx = -3.4e38f;
  for (int c = 0; c < C; ++c) {
    s  += A[((long)(n*C + c))*HW + p];
    float bv = B[((long)(n*C + c))*HW + p];
    mx = fmaxf(mx, bv);
  }
  Si[((long)(n*2 + 0))*HW + p] = s / (float)C;
  Si[((long)(n*2 + 1))*HW + p] = mx;
}
__global__ void k_conv3x3_s2_sig(const float* __restrict__ Si, const float* __restrict__ Wt,
                                 float* __restrict__ Al, int N, int H, int W)
{
  long idx = (long)blockIdx.x*blockDim.x + threadIdx.x;
  long total = (long)N*H*W;
  if (idx >= total) return;
  int w = idx % W; int h = (idx / W) % H; int n = idx / ((long)H*W);
  float s = 0.f;
  for (int ci = 0; ci < 2; ++ci) {
    const float* xc = Si + (long)(n*2 + ci)*H*W;
    for (int dy = -1; dy <= 1; ++dy) {
      int y = h + dy; if ((unsigned)y >= (unsigned)H) continue;
      for (int dx = -1; dx <= 1; ++dx) {
        int x = w + dx; if ((unsigned)x >= (unsigned)W) continue;
        s += xc[(long)y*W + x] * Wt[(ci*3 + (dy+1))*3 + (dx+1)];
      }
    }
  }
  Al[idx] = sigm(s);
}
__global__ void k_mix(const float* __restrict__ A, const float* __restrict__ B,
                      const float* __restrict__ Al, float* __restrict__ O,
                      int N, int C, int HW)
{
  long idx = (long)blockIdx.x*blockDim.x + threadIdx.x;
  long total = (long)N*C*HW;
  if (idx >= total) return;
  int p = idx % HW; int n = idx / ((long)C*HW);
  float al = Al[(long)n*HW + p];
  O[idx] = A[idx]*al + B[idx]*(1.f - al);
}
__global__ void k_gap(const float* __restrict__ X, float* __restrict__ Y,
                      int N, int C, int HW)
{
  long idx = (long)blockIdx.x*blockDim.x + threadIdx.x;
  if (idx >= (long)N*C) return;
  const float* xc = X + idx*(long)HW;
  float s = 0.f;
  for (int p = 0; p < HW; ++p) s += xc[p];
  Y[idx] = s / (float)HW;
}
__global__ void k_eca_sig(const float* __restrict__ Yg, const float* __restrict__ Kk,
                          float* __restrict__ Sc, int N, int C)
{
  long idx = (long)blockIdx.x*blockDim.x + threadIdx.x;
  if (idx >= (long)N*C) return;
  int c = idx % C; int n = idx / C;
  const float* yn = Yg + (long)n*C;
  float s = 0.f;
  for (int i = 0; i < 3; ++i) {
    int cc = c + i - 1;
    if ((unsigned)cc < (unsigned)C) s += yn[cc] * Kk[i];
  }
  Sc[idx] = sigm(s);
}
__global__ void k_scale_ch(const float* __restrict__ X, const float* __restrict__ Sc,
                           float* __restrict__ Y, int N, int C, int HW)
{
  long idx = (long)blockIdx.x*blockDim.x + threadIdx.x;
  long total = (long)N*C*HW;
  if (idx >= total) return;
  long nc = idx / HW;
  Y[idx] = X[idx] * Sc[nc];
}

// ===========================================================================
// host orchestration
// ===========================================================================
struct BN   { const float *g,*b,*m,*v; };
struct CADW { const float *r,*wh,*ww; };
struct CAp  { const float* c1; BN bn1; const float* a1; const float* ch; const float* cw; };
struct DGp  { const float *wh,*ww; BN bn; };
struct PFCU { CADW m,l; const float* pw; BN bnf; DGp dg; CAp ca; const float* act; };
struct ENC  { PFCU pfcu; BN bn; const float* pw; BN bn2; const float* act; };
struct POOLP{ const float* w; BN bn; const float* a; };
struct PROJ { const float* w; BN bn; const float* a; };
struct UAFMp{ const float* sp; const float* eca; };

static inline int nblk(long n, int b) { return (int)((n + b - 1) / b); }

extern "C" void kernel_launch(void* const* d_in, const int* in_sizes, int n_in,
                              void* d_out, int out_size, void* d_ws, size_t ws_size,
                              hipStream_t stream)
{
  (void)in_sizes; (void)n_in; (void)out_size; (void)ws_size;
  int ii = 0;
  auto nxt = [&]() -> const float* { return (const float*)d_in[ii++]; };
  auto getBN   = [&]() { BN r;   r.g=nxt(); r.b=nxt(); r.m=nxt(); r.v=nxt(); return r; };
  auto getCADW = [&]() { CADW r; r.r=nxt(); r.wh=nxt(); r.ww=nxt(); return r; };
  auto getCA   = [&]() { CAp r;  r.c1=nxt(); r.bn1=getBN(); r.a1=nxt(); r.ch=nxt(); r.cw=nxt(); return r; };
  auto getDG   = [&]() { DGp r;  r.wh=nxt(); r.ww=nxt(); r.bn=getBN(); return r; };
  auto getPFCU = [&]() { PFCU r; r.m=getCADW(); r.l=getCADW(); r.pw=nxt(); r.bnf=getBN();
                         r.dg=getDG(); r.ca=getCA(); r.act=nxt(); return r; };
  auto getENC  = [&]() { ENC r;  r.pfcu=getPFCU(); r.bn=getBN(); r.pw=nxt(); r.bn2=getBN(); r.act=nxt(); return r; };
  auto getPROJ = [&]() { PROJ r; r.w=nxt(); r.bn=getBN(); r.a=nxt(); return r; };
  auto getUAFM = [&]() { UAFMp r; r.sp=nxt(); r.eca=nxt(); return r; };

  const float* Xin = nxt();                 // x (4,3,512,512)
  const float* in_w = nxt();
  const float* in_b = nxt();
  ENC e1 = getENC(), e2 = getENC(), e3 = getENC(), e4 = getENC();
  POOLP pools[3];
  for (int i = 0; i < 3; ++i) { pools[i].w = nxt(); pools[i].bn = getBN(); pools[i].a = nxt(); }
  const float* fw = nxt(); BN fbn = getBN(); const float* fa = nxt();
  const float* dw = nxt(); BN bnr = getBN(); CAp b5ca = getCA();
  PROJ p16 = getPROJ(), p8 = getPROJ(); UAFMp u8 = getUAFM();
  PROJ p84 = getPROJ(), p4 = getPROJ(); UAFMp u4 = getUAFM();
  const float* cls_w = nxt(); const float* cls_b = nxt();

  // ---- workspace arena (floats) ----
  float* ws = (float*)d_ws;
  const long S0 = 4L*16*512*512;            // 16.78M floats
  float* B0 = ws;        float* B1 = B0 + S0;
  float* B2 = B1 + S0;   float* B3 = B2 + S0;
  float* SK3 = B3 + S0;                     // s3: 4*64*128*128
  float* SK4 = SK3 + 4L*64*128*128;         // s4: 4*128*64*64
  float* SM  = SK4 + 4L*128*64*64;          // small-tensor arena

  auto conv1x1 = [&](const float* Xp, int XHW, int xoff, const float* Wp,
                     float* Yp, int YHW, int yoff, int Cin, int Cout, int HW,
                     int mode, const BN* bn, const float* pa, const float* bias) {
    long Npix = (long)NB*HW;
    dim3 grid((unsigned)((Npix + TILE_PX - 1)/TILE_PX), (unsigned)((Cout + TILE_CO - 1)/TILE_CO));
    k_conv1x1<<<grid, 256, 0, stream>>>(Xp, XHW, xoff, Wp,
        bn ? bn->g : nullptr, bn ? bn->b : nullptr,
        bn ? bn->m : nullptr, bn ? bn->v : nullptr,
        pa, bias, Yp, YHW, yoff, Cin, Cout, HW, (int)Npix, mode);
  };
  auto bn_act = [&](const float* Xp, float* Yp, const BN& bn, const float* pa,
                    int C, long HW, int mode) {
    long total = (long)NB*C*HW;
    k_bn_act<<<nblk(total,256), 256, 0, stream>>>(Xp, Yp, bn.g, bn.b, bn.m, bn.v,
                                                  pa, C, (int)HW, total, mode);
  };
  auto ca_run = [&](const float* Xp, int C, int H, int W, const CAp& ca, float* Yp) {
    int L = H + W;
    float* ycat = SM;
    float* y8   = ycat + (long)NB*C*L;
    float* ah   = y8   + (long)NB*8*L;
    float* aw   = ah   + (long)NB*C*H;
    long t1 = (long)NB*C*L;
    k_ca_pool<<<nblk(t1,256), 256, 0, stream>>>(Xp, ycat, NB, C, H, W);
    conv1x1(ycat, L, 0, ca.c1, y8, L, 0, C, 8, L, M_BN|M_PRELU, &ca.bn1, ca.a1, nullptr);
    conv1x1(y8, L, 0, ca.ch, ah, H, 0, 8, C, H, M_SIG, nullptr, nullptr, nullptr);
    conv1x1(y8, L, H, ca.cw, aw, W, 0, 8, C, W, M_SIG, nullptr, nullptr, nullptr);
    long t2 = (long)NB*C*H*W;
    k_ca_apply<<<nblk(t2,256), 256, 0, stream>>>(Xp, ah, aw, Yp, NB, C, H, W);
  };
  auto run_enc = [&](const float* Xi, int Ci, int R, const ENC& e, int K,
                     float* t1, float* t2, float* skipb, float* outb) {
    long HW = (long)R*R; long tot = (long)NB*Ci*HW;
    k_cadw2<<<nblk(tot,256), 256, 0, stream>>>(Xi, e.pfcu.m.r, e.pfcu.m.wh, e.pfcu.m.ww,
        e.pfcu.l.r, e.pfcu.l.wh, e.pfcu.l.ww, t1, NB, Ci, R, R, K);
    conv1x1(t1, (int)HW, 0, e.pfcu.pw, t2, (int)HW, 0, Ci, Ci, (int)HW,
            M_BN, &e.pfcu.bnf, nullptr, nullptr);
    k_dg_add_prelu<<<nblk(tot,256), 256, 0, stream>>>(Xi, t2, e.pfcu.dg.wh, e.pfcu.dg.ww,
        e.pfcu.dg.bn.g, e.pfcu.dg.bn.b, e.pfcu.dg.bn.m, e.pfcu.dg.bn.v,
        e.pfcu.act, t1, NB, Ci, R, R);
    ca_run(t1, Ci, R, R, e.pfcu.ca, t2);
    bn_act(t2, skipb, e.bn, nullptr, Ci, HW, M_BN);
    conv1x1(skipb, (int)HW, 0, e.pw, t1, (int)HW, 0, Ci, Ci, (int)HW,
            0, nullptr, nullptr, nullptr);
    int Ro = R/2; long HWo = (long)Ro*Ro; long pt = (long)NB*Ci*HWo;
    k_maxpool2<<<nblk(pt,256), 256, 0, stream>>>(skipb, outb, NB, Ci, R, R, 2*Ci, 0);
    k_maxpool2<<<nblk(pt,256), 256, 0, stream>>>(t1,    outb, NB, Ci, R, R, 2*Ci, Ci);
    bn_act(outb, outb, e.bn2, e.act, 2*Ci, HWo, M_BN|M_PRELU);
  };
  auto run_uafm = [&](const float* xup, const float* xsk, int C, int H, int W,
                      const UAFMp& u, float* outp) {
    long HW = (long)H*W;
    float* si    = SM;
    float* alpha = si + (long)NB*2*HW;
    float* gp    = alpha + (long)NB*HW;
    float* sc    = gp + (long)NB*C;
    long st = (long)NB*HW;
    k_chan_reduce<<<nblk(st,256), 256, 0, stream>>>(xup, xsk, si, NB, C, (int)HW);
    k_conv3x3_s2_sig<<<nblk(st,256), 256, 0, stream>>>(si, u.sp, alpha, NB, H, W);
    long mt = (long)NB*C*HW;
    k_mix<<<nblk(mt,256), 256, 0, stream>>>(xup, xsk, alpha, outp, NB, C, (int)HW);
    k_gap<<<nblk((long)NB*C,64), 64, 0, stream>>>(outp, gp, NB, C, (int)HW);
    k_eca_sig<<<nblk((long)NB*C,64), 64, 0, stream>>>(gp, u.eca, sc, NB, C);
    k_scale_ch<<<nblk(mt,256), 256, 0, stream>>>(outp, sc, outp, NB, C, (int)HW);
  };

  // ---- forward ----
  long it = (long)NB*16*512*512;
  k_conv3x3_in<<<nblk(it,256), 256, 0, stream>>>(Xin, in_w, in_b, B0, NB, 512, 512);

  run_enc(B0, 16, 512, e1, 5, B1, B2, B2,  B3);   // -> 32 @256 in B3
  run_enc(B3, 32, 256, e2, 5, B0, B1, B1,  B2);   // -> 64 @128 in B2
  run_enc(B2, 64, 128, e3, 7, B0, B1, SK3, B1);   // skip s3 -> SK3; -> 128 @64 in B1
  run_enc(B1, 128, 64, e4, 7, B0, B2, SK4, B2);   // skip s4 -> SK4; -> 256 @32 in B2

  // bottleneck: input B2 (256 @32)
  {
    const int H = 32, W = 32; const long HW = 1024;
    float* feats = B0; const int CF = 352;
    long ct = (long)NB*256*HW;
    k_copy_ch<<<nblk(ct,256), 256, 0, stream>>>(B2, feats, NB, 256, (int)HW, CF, 0);
    float* pl = SM;
    float* yv = SM + (long)NB*256*16;
    const int os[3] = {1, 2, 4};
    for (int i = 0; i < 3; ++i) {
      int o = os[i];
      long pt = (long)NB*256*o*o;
      k_adaptive_avg<<<nblk(pt,256), 256, 0, stream>>>(B2, pl, NB, 256, H, W, o);
      conv1x1(pl, o*o, 0, pools[i].w, yv, o*o, 0, 256, 32, o*o,
              M_BN|M_PRELU, &pools[i].bn, pools[i].a, nullptr);
      long rt = (long)NB*32*HW;
      k_resize<<<nblk(rt,256), 256, 0, stream>>>(yv, NB, 32, o, o, feats, H, W, CF, 256 + 32*i);
    }
    float* fused = B1;
    conv1x1(feats, (int)HW, 0, fw, fused, (int)HW, 0, 352, 256, (int)HW,
            M_BN|M_PRELU, &fbn, fa, nullptr);
    float* t = B3;
    long tt = (long)NB*256*HW;
    k_dw5_bn_add<<<nblk(tt,256), 256, 0, stream>>>(fused, dw, bnr.g, bnr.b, bnr.m, bnr.v,
                                                   t, NB, 256, H, W);
    ca_run(t, 256, H, W, b5ca, B0);               // bottleneck out -> B0 (256 @32)
  }

  // decoder
  float* p16o = B1;
  conv1x1(B0, 1024, 0, p16.w, p16o, 1024, 0, 256, 128, 1024, M_BN|M_PRELU, &p16.bn, p16.a, nullptr);
  float* x16u = B2;
  k_resize<<<nblk((long)NB*128*4096,256), 256, 0, stream>>>(p16o, NB, 128, 32, 32, x16u, 64, 64, 128, 0);
  float* f8p = B1;
  conv1x1(SK4, 4096, 0, p8.w, f8p, 4096, 0, 128, 128, 4096, M_BN|M_PRELU, &p8.bn, p8.a, nullptr);
  float* fuse8 = B3;
  run_uafm(x16u, f8p, 128, 64, 64, u8, fuse8);
  float* p84o = B1;
  conv1x1(fuse8, 4096, 0, p84.w, p84o, 4096, 0, 128, 8, 4096, M_BN|M_PRELU, &p84.bn, p84.a, nullptr);
  float* x84u = B2;
  k_resize<<<nblk((long)NB*8*16384,256), 256, 0, stream>>>(p84o, NB, 8, 64, 64, x84u, 128, 128, 8, 0);
  float* f4p = B1;
  conv1x1(SK3, 16384, 0, p4.w, f4p, 16384, 0, 64, 8, 16384, M_BN|M_PRELU, &p4.bn, p4.a, nullptr);
  float* fuse4 = B3;
  run_uafm(x84u, f4p, 8, 128, 128, u4, fuse4);
  float* outr = B0;
  k_resize<<<nblk((long)NB*8*262144,256), 256, 0, stream>>>(fuse4, NB, 8, 128, 128, outr, 512, 512, 8, 0);
  conv1x1(outr, 262144, 0, cls_w, (float*)d_out, 262144, 0, 8, 1, 262144,
          M_BIAS, nullptr, nullptr, cls_b);
}
